// AnswerHead_12721693131030
// MI455X (gfx1250) — compile-verified, matrix operand order checked by reading, not run on hardware
//
#include <hip/hip_runtime.h>

typedef __attribute__((ext_vector_type(16))) __bf16 v16bf;
typedef __attribute__((ext_vector_type(8)))  __bf16 v8bf;
typedef __attribute__((ext_vector_type(8)))  float  v8f;
typedef __attribute__((ext_vector_type(4)))  float  v4f;

#define D_DIM 768
#define V_DIM 30000
#define M_DIM 640              // L*B*T = 2*16*20
#define TILE_V 64
#define LDS_STRIDE (D_DIM + 8) // pad 8 bf16 (16B) to stagger LDS banks

// ---------------------------------------------------------------------------
// Fragment loaders. WMMA bf16 operand layout: element e of lane L holds
// K = k0 + 16*(e/8) + 8*(L/16) + (e%8)  -> two contiguous 8-element segments.
// ---------------------------------------------------------------------------

// fp32 source, convert to bf16 on the fly. NT=true -> non-temporal loads
// (used for the single-use HBM-streamed vocab rows).
template <bool NT>
__device__ __forceinline__ v16bf load16_f32_bf16(const float* __restrict__ rowp,
                                                 int k0, int half) {
    const v4f* p0 = (const v4f*)(rowp + k0 + 8 * half);
    const v4f* p1 = (const v4f*)(rowp + k0 + 16 + 8 * half);
    v4f x0, x1, y0, y1;
    if constexpr (NT) {
        x0 = __builtin_nontemporal_load(p0);
        x1 = __builtin_nontemporal_load(p0 + 1);
        y0 = __builtin_nontemporal_load(p1);
        y1 = __builtin_nontemporal_load(p1 + 1);
    } else {
        x0 = p0[0]; x1 = p0[1]; y0 = p1[0]; y1 = p1[1];
    }
    v16bf r;
    r[0]  = (__bf16)x0.x; r[1]  = (__bf16)x0.y; r[2]  = (__bf16)x0.z; r[3]  = (__bf16)x0.w;
    r[4]  = (__bf16)x1.x; r[5]  = (__bf16)x1.y; r[6]  = (__bf16)x1.z; r[7]  = (__bf16)x1.w;
    r[8]  = (__bf16)y0.x; r[9]  = (__bf16)y0.y; r[10] = (__bf16)y0.z; r[11] = (__bf16)y0.w;
    r[12] = (__bf16)y1.x; r[13] = (__bf16)y1.y; r[14] = (__bf16)y1.z; r[15] = (__bf16)y1.w;
    return r;
}

// bf16 source (pre-converted global tensors, or the LDS-staged VC tile):
// two 16-byte loads, no VALU conversion work.
__device__ __forceinline__ v16bf load16_bf16(const __bf16* rowp, int k0, int half) {
    v8bf s0 = *(const v8bf*)(rowp + k0 + 8 * half);
    v8bf s1 = *(const v8bf*)(rowp + k0 + 16 + 8 * half);
    v16bf r;
#pragma unroll
    for (int i = 0; i < 8; ++i) { r[i] = s0[i]; r[8 + i] = s1[i]; }
    return r;
}

// Row fragment from a [rows, D_DIM] row-major matrix, bf16 or fp32 source.
template <bool BF>
__device__ __forceinline__ v16bf load_row_frag(const void* __restrict__ base,
                                               int row, int k0, int half) {
    if constexpr (BF) {
        return load16_bf16((const __bf16*)base + (size_t)row * D_DIM, k0, half);
    } else {
        return load16_f32_bf16<false>((const float*)base + (size_t)row * D_DIM, k0, half);
    }
}

#define WMMA_BF16(A, B, C) \
    __builtin_amdgcn_wmma_f32_16x16x32_bf16(false, (A), false, (B), (short)0, (C), false, false)

// ---------------------------------------------------------------------------
// Prologue: fp32 -> bf16 repack (8 elements per thread).
// ---------------------------------------------------------------------------
__global__ __launch_bounds__(256)
void f32_to_bf16_kernel(const float* __restrict__ src, __bf16* __restrict__ dst, int n8) {
    int i = blockIdx.x * blockDim.x + threadIdx.x;
    if (i < n8) {
        const v4f* p = (const v4f*)(src + (size_t)i * 8);
        v4f a = p[0];
        v4f b = p[1];
        v8bf r;
        r[0] = (__bf16)a.x; r[1] = (__bf16)a.y; r[2] = (__bf16)a.z; r[3] = (__bf16)a.w;
        r[4] = (__bf16)b.x; r[5] = (__bf16)b.y; r[6] = (__bf16)b.z; r[7] = (__bf16)b.w;
        *(v8bf*)(dst + (size_t)i * 8) = r;
    }
}

// ---------------------------------------------------------------------------
// Fused: per V-tile, VC tile = vocab @ W^T + b  (kept bf16 in LDS, never HBM),
// then out tile = answer @ VC^T.
// PRECONV: answer/W already bf16 in workspace.
// ---------------------------------------------------------------------------
template <bool PRECONV>
__global__ __launch_bounds__(256)
void answer_head_fused(const void*  __restrict__ A,      // [640, 768]   bf16 or fp32
                       const float* __restrict__ vocab,  // [30000, 768] fp32 (streamed)
                       const void*  __restrict__ W,      // [768, 768]   bf16 or fp32
                       const float* __restrict__ bias,   // [768]
                       float* __restrict__ out)          // [640, 30000]
{
    __shared__ __bf16 vc[TILE_V * LDS_STRIDE];  // bf16 VC tile: 64 x 768 (+pad)

    const int tid  = threadIdx.x;
    const int wave = tid >> 5;
    const int lane = tid & 31;
    const int half = lane >> 4;
    const int l16  = lane & 15;

    const int v0 = blockIdx.x * TILE_V;

    // ---- Phase 1: vc[r, j] = bf16(vocab[v0+r,:] . W[j,:] + b[j]) ------------
    {
        const int rtile = wave & 3;   // 4 row-tiles of 16 rows
        const int chalf = wave >> 2;  // 2 column halves of 384 cols

        int vrow = v0 + rtile * 16 + l16;
        if (vrow >= V_DIM) vrow = V_DIM - 1;  // clamp tail loads (stores guarded)
        const float* arow = vocab + (size_t)vrow * D_DIM;

        // Pull the lane's whole 3KB vocab row toward the WGP up front:
        // 24 unconditional global_prefetch_b8 (one per 128B line), no
        // control flow in the hot loops.
#pragma unroll
        for (int p = 0; p < D_DIM; p += 32)
            __builtin_prefetch(arow + p, 0, 1);

#pragma unroll 1
        for (int jg = 0; jg < 6; ++jg) {
            const int jbase = chalf * 384 + jg * 64;
            const int j0 = jbase + l16;

            v8f c0 = {}, c1 = {}, c2 = {}, c3 = {};
#pragma unroll 2
            for (int k0 = 0; k0 < D_DIM; k0 += 32) {
                v16bf a  = load16_f32_bf16<true>(arow, k0, half);
                v16bf b0 = load_row_frag<PRECONV>(W, j0 +  0, k0, half);
                v16bf b1 = load_row_frag<PRECONV>(W, j0 + 16, k0, half);
                v16bf b2 = load_row_frag<PRECONV>(W, j0 + 32, k0, half);
                v16bf b3 = load_row_frag<PRECONV>(W, j0 + 48, k0, half);
                c0 = WMMA_BF16(a, b0, c0);
                c1 = WMMA_BF16(a, b1, c1);
                c2 = WMMA_BF16(a, b2, c2);
                c3 = WMMA_BF16(a, b3, c3);
            }

            // bias add + bf16 store into the LDS tile.
            // D layout: VGPR r of lane L -> row = r + 8*(L/16), col = L%16.
#define STORE_VC(ACC, JT)                                                    \
            {                                                                \
                const int col = jbase + (JT) * 16 + l16;                     \
                const float bj = bias[col];                                  \
                _Pragma("unroll")                                            \
                for (int r = 0; r < 8; ++r) {                                \
                    const int row = rtile * 16 + r + 8 * half;               \
                    vc[row * LDS_STRIDE + col] = (__bf16)(ACC[r] + bj);      \
                }                                                            \
            }
            STORE_VC(c0, 0)
            STORE_VC(c1, 1)
            STORE_VC(c2, 2)
            STORE_VC(c3, 3)
#undef STORE_VC
        }
    }

    __syncthreads();

    // ---- Phase 2: out[m, v0+c] = answer[m,:] . vc[c,:] ----------------------
    {
        const int ct    = wave & 3;   // 16-column slice of the V tile
        const int mhalf = wave >> 2;  // half of the 640 M rows
        const bool col_valid = (v0 + ct * 16) < V_DIM;  // V%16==0 -> tile test

        const __bf16* brow = vc + (size_t)(ct * 16 + l16) * LDS_STRIDE;

#pragma unroll 1
        for (int mg = 0; mg < 5; ++mg) {
            const int m0 = mhalf * 320 + mg * 64;
            const int mr = m0 + l16;

            v8f c0 = {}, c1 = {}, c2 = {}, c3 = {};
#pragma unroll 2
            for (int k0 = 0; k0 < D_DIM; k0 += 32) {
                v16bf b  = load16_bf16(brow, k0, half);  // LDS, reused 4x
                v16bf f0 = load_row_frag<PRECONV>(A, mr +  0, k0, half);
                v16bf f1 = load_row_frag<PRECONV>(A, mr + 16, k0, half);
                v16bf f2 = load_row_frag<PRECONV>(A, mr + 32, k0, half);
                v16bf f3 = load_row_frag<PRECONV>(A, mr + 48, k0, half);
                c0 = WMMA_BF16(f0, b, c0);
                c1 = WMMA_BF16(f1, b, c1);
                c2 = WMMA_BF16(f2, b, c2);
                c3 = WMMA_BF16(f3, b, c3);
            }

            if (col_valid) {
                const int vcol = v0 + ct * 16 + l16;
                // Output is written once and never re-read: non-temporal
                // stores keep the 77MB result stream out of L2.
#define STORE_OUT(ACC, MT)                                                   \
                {                                                            \
                    _Pragma("unroll")                                        \
                    for (int r = 0; r < 8; ++r) {                            \
                        const int m = m0 + (MT) * 16 + r + 8 * half;         \
                        __builtin_nontemporal_store(                         \
                            ACC[r], &out[(size_t)m * V_DIM + vcol]);         \
                    }                                                        \
                }
                STORE_OUT(c0, 0)
                STORE_OUT(c1, 1)
                STORE_OUT(c2, 2)
                STORE_OUT(c3, 3)
#undef STORE_OUT
            }
        }
    }
}

extern "C" void kernel_launch(void* const* d_in, const int* in_sizes, int n_in,
                              void* d_out, int out_size, void* d_ws, size_t ws_size,
                              hipStream_t stream) {
    (void)in_sizes; (void)n_in; (void)out_size;
    const float* answer = (const float*)d_in[0];  // [2,16,20,768] -> [640,768]
    const float* vocab  = (const float*)d_in[1];  // [30000,768]
    const float* W      = (const float*)d_in[2];  // [768,768]
    const float* bias   = (const float*)d_in[3];  // [768]
    float* out = (float*)d_out;                   // [640,30000]

    const dim3 grid((V_DIM + TILE_V - 1) / TILE_V);  // 469 blocks

    const size_t needA = (size_t)M_DIM * D_DIM * sizeof(__bf16);  // 983,040 B
    const size_t needW = (size_t)D_DIM * D_DIM * sizeof(__bf16);  // 1,179,648 B

    if (ws_size >= needA + needW) {
        __bf16* Ab = (__bf16*)d_ws;
        __bf16* Wb = (__bf16*)((char*)d_ws + needA);
        const int nA8 = (M_DIM * D_DIM) / 8;  // 61,440
        const int nW8 = (D_DIM * D_DIM) / 8;  // 73,728
        f32_to_bf16_kernel<<<(nA8 + 255) / 256, 256, 0, stream>>>(answer, Ab, nA8);
        f32_to_bf16_kernel<<<(nW8 + 255) / 256, 256, 0, stream>>>(W, Wb, nW8);
        answer_head_fused<true><<<grid, 256, 0, stream>>>(Ab, vocab, Wb, bias, out);
    } else {
        // Workspace too small: self-contained path, convert W/answer on the fly.
        answer_head_fused<false><<<grid, 256, 0, stream>>>(answer, vocab, W, bias, out);
    }
}
